// LocalContextAttentionBlock_6828998000689
// MI455X (gfx1250) — matrix operand order repaired and checked by reading, not compiled, not run on hardware
//
#include <hip/hip_runtime.h>
#include <hip/hip_bf16.h>
#include <math.h>

// ---------------------------------------------------------------------------
// Problem constants (fixed by the reference):
//   B=2, C=O=128, H=W=180, HW=32400, 9x9 window, pad 4.
// ---------------------------------------------------------------------------
#define BATCH   2
#define CH      128
#define HH      180
#define WW      180
#define HWSZ    (HH * WW)          // 32400, multiple of 16
#define NTILES  (HWSZ / 16)        // 2025
#define MTILES  (CH / 16)          // 8
#define EPS     1e-5f

// LDS staging of X chunk: 128 K-rows x 16 pixels, row stride 17 floats
// (1-DWORD pad per 16 DWORDs, produced by the TDM pad feature) so the two
// lane-halves of a wave read disjoint LDS banks.
#define BROW    17
#define BSTAGE  (CH * BROW)        // 2176 floats = 8704 B

typedef __attribute__((ext_vector_type(16))) _Float16 v16h;
typedef __attribute__((ext_vector_type(8)))  float    v8f;
typedef unsigned int u32x4 __attribute__((ext_vector_type(4)));
typedef int          i32x8 __attribute__((ext_vector_type(8)));
typedef int          i32x4 __attribute__((ext_vector_type(4)));

// ---------------------------------------------------------------------------
// Fused 1x1 conv + BN + ReLU as a WMMA GEMM:
//   Y[o, n] = relu( scale[o] * sum_c W[o,c] * X[c,n] + shift[o] )
// Workgroup = 8 waves = all 8 output-channel tiles of one 16-pixel column
// group. The shared X chunk [K=128][N=16] is DMA'd into LDS once per block
// by the Tensor Data Mover (tensor_load_to_lds + s_wait_tensorcnt), then
// each wave runs 4x v_wmma_f32_16x16x32_f16 over K.
// ---------------------------------------------------------------------------
__global__ __launch_bounds__(256)
void proj_gemm_kernel(const float* __restrict__ X,
                      const float* __restrict__ Wm,   // [O=128][C=128]
                      const float* __restrict__ bn_g,
                      const float* __restrict__ bn_b,
                      const float* __restrict__ bn_m,
                      const float* __restrict__ bn_v,
                      float* __restrict__ Y)
{
    __shared__ float ldsb[BSTAGE];

    const int ntile = blockIdx.x % NTILES;
    const int batch = blockIdx.x / NTILES;
    const int mtile = threadIdx.x >> 5;            // wave id = M tile
    const int lane  = threadIdx.x & 31;
    const int khalf = lane >> 4;
    const int l16   = lane & 15;
    const int mbase = mtile * 16;
    const int n0    = ntile * 16;
    const int ncol  = n0 + l16;

    const float* Xg = X + (size_t)batch * CH * HWSZ + n0;  // tile origin

#if __has_builtin(__builtin_amdgcn_tensor_load_to_lds)
    // ---- TDM stage: 2D tile, tile_dim0=16 elems (64B rows), tile_dim1=128
    // rows, dim0 stride = HW, 4-byte elements, LDS pad 1 DWORD / 16 DWORDs.
    if (threadIdx.x < 32) {
        const unsigned long long ga = (unsigned long long)(uintptr_t)Xg;
        u32x4 g0;
        g0.x = 1u;                                  // count = 1 (valid D#)
        g0.y = (unsigned int)(uintptr_t)&ldsb[0];   // lds_addr (bytes)
        g0.z = (unsigned int)ga;                    // global_addr[31:0]
        g0.w = (unsigned int)(ga >> 32) | (2u << 30);  // addr hi | type=2
        i32x8 g1;
        g1[0] = (int)((2u << 16)        // data_size = 2 -> 4 bytes
                    | (1u << 20)        // pad_enable
                    | (3u << 22));      // pad_interval code 3 = 16 DWORDs
                                        // pad_amount  code 0 = 1 DWORD
        g1[1] = (int)(0x7E90u << 16);   // tensor_dim0 = 32400 (lo16 @ [31:16])
        g1[2] = (int)(128u << 16);      // tensor_dim0 hi16=0 | tensor_dim1=128
        g1[3] = (int)(16u << 16);       // tensor_dim1 hi16=0 | tile_dim0=16
        g1[4] = (int)128;               // tile_dim1=128 | tile_dim2=0
        g1[5] = (int)HWSZ;              // tensor_dim0_stride lo32
        g1[6] = 0;                      // stride hi | tensor_dim1_stride lo
        g1[7] = 0;
        i32x4 g2 = {0, 0, 0, 0};
        i32x4 g3 = {0, 0, 0, 0};
        i32x8 g4 = {0, 0, 0, 0, 0, 0, 0, 0};
        __builtin_amdgcn_tensor_load_to_lds(g0, g1, g2, g3, g4, 0);
        __builtin_amdgcn_s_wait_tensorcnt(0);
    }
#else
    // ---- Fallback: cooperative staging with padded rows.
    for (int i = threadIdx.x; i < CH * 16; i += 256) {
        const int k = i >> 4, n = i & 15;
        ldsb[k * BROW + n] = Xg[(size_t)k * HWSZ + n];
    }
#endif
    __syncthreads();

    // A rows: output channel m = mbase + l16 (both lane-halves same row,
    // different K segments per the 16-bit A-matrix layout).
    const float* Arow = Wm + (size_t)(mbase + l16) * CH;

    v8f acc = {};
#pragma unroll
    for (int kk = 0; kk < 4; ++kk) {
        const int k0 = kk * 32;

        // ---- A fragment: 16x32 f16.  lane l (m = l%16, kh = l/16):
        //   a[0..7]  = W[m][k0 + kh*8 + 0..7]
        //   a[8..15] = W[m][k0 + 16 + kh*8 + 0..7]
        v16h av;
        {
            const float* ap0 = Arow + k0 + khalf * 8;
            const float* ap1 = Arow + k0 + 16 + khalf * 8;
            const float4 a00 = *(const float4*)(ap0);
            const float4 a01 = *(const float4*)(ap0 + 4);
            const float4 a10 = *(const float4*)(ap1);
            const float4 a11 = *(const float4*)(ap1 + 4);
            av[0]  = (_Float16)a00.x; av[1]  = (_Float16)a00.y;
            av[2]  = (_Float16)a00.z; av[3]  = (_Float16)a00.w;
            av[4]  = (_Float16)a01.x; av[5]  = (_Float16)a01.y;
            av[6]  = (_Float16)a01.z; av[7]  = (_Float16)a01.w;
            av[8]  = (_Float16)a10.x; av[9]  = (_Float16)a10.y;
            av[10] = (_Float16)a10.z; av[11] = (_Float16)a10.w;
            av[12] = (_Float16)a11.x; av[13] = (_Float16)a11.y;
            av[14] = (_Float16)a11.z; av[15] = (_Float16)a11.w;
        }

        // ---- B fragment from LDS: 32x16 f16.  lane l (n = l%16, kh = l/16):
        //   element t = Xstage[k0 + (t/8)*16 + kh*8 + t%8][n]   (row stride 17 floats)
        v16h bv;
#pragma unroll
        for (int t = 0; t < 16; ++t)
            bv[t] = (_Float16)ldsb[(k0 + (t >> 3) * 16 + khalf * 8 + (t & 7)) * BROW + l16];

        acc = __builtin_amdgcn_wmma_f32_16x16x32_f16(
                  false, av, false, bv, (short)0, acc, false, false);
    }

    // ---- Epilogue: BN fold + ReLU.  D element r -> M = r + 8*khalf, N = l16.
    float* Yb = Y + (size_t)batch * CH * HWSZ;
#pragma unroll
    for (int r = 0; r < 8; ++r) {
        const int o   = mbase + r + 8 * khalf;
        const float s = bn_g[o] * rsqrtf(bn_v[o] + EPS);
        const float t = bn_b[o] - bn_m[o] * s;
        float y = fmaf(acc[r], s, t);
        y = y > 0.f ? y : 0.f;
        Yb[(size_t)o * HWSZ + ncol] = y;
    }
}

// ---------------------------------------------------------------------------
// Fused local 9x9 attention: sims -> softmax -> weighted sum.
// Block = 256 threads = 4(H) x 64(W) pixel tile. Per 4-channel chunk the
// (4+8) x (64+8) halo of key (then value) is staged to LDS as float4-per-
// pixel (ds_load_b128 reads), the 81 logits live in registers.
// (Manual staging, not TDM: border tiles need negative halo offsets with
// zero-fill, which the TDM descriptor cannot express.)
// ---------------------------------------------------------------------------
#define TW      64
#define THT     4
#define HALO_W  (TW + 8)                 // 72
#define HALO_H  (THT + 8)                // 12
#define TILE_E  (HALO_W * HALO_H)        // 864 pixels, *4 ch = 3456 floats

__global__ __launch_bounds__(256)
void local_attn_kernel(const float* __restrict__ q,
                       const float* __restrict__ key,
                       const float* __restrict__ val,
                       float* __restrict__ out)
{
    __shared__ float tile[TILE_E * 4];   // 13.8 KB

    const int tid = threadIdx.x;
    const int tw  = tid & (TW - 1);
    const int th  = tid >> 6;
    const int w0  = blockIdx.x * TW;
    const int h0  = blockIdx.y * THT;
    const int b   = blockIdx.z;
    const int w   = w0 + tw;
    const int h   = h0 + th;
    const bool valid = (w < WW);         // h always < 180 (45*4)

    const size_t bbase = (size_t)b * CH * HWSZ;

    float sims[81];
#pragma unroll
    for (int o = 0; o < 81; ++o) sims[o] = 0.f;

    // ---------------- Pass 1: similarity logits ----------------
    for (int c0 = 0; c0 < CH; c0 += 4) {
        __syncthreads();
        for (int i = tid; i < TILE_E * 4; i += 256) {
            const int col = i % HALO_W;                 // coalesced along W
            const int r   = (i / HALO_W) % HALO_H;
            const int cc  = i / (HALO_W * HALO_H);
            const int gh  = h0 + r - 4;
            const int gw  = w0 + col - 4;
            float kv = 0.f;
            if (gh >= 0 && gh < HH && gw >= 0 && gw < WW)
                kv = key[bbase + (size_t)(c0 + cc) * HWSZ + gh * WW + gw];
            tile[(r * HALO_W + col) * 4 + cc] = kv;
        }
        __syncthreads();

        float q0 = 0.f, q1 = 0.f, q2 = 0.f, q3 = 0.f;
        if (valid) {
            const size_t qi = bbase + (size_t)c0 * HWSZ + h * WW + w;
            q0 = q[qi];
            q1 = q[qi + HWSZ];
            q2 = q[qi + 2 * HWSZ];
            q3 = q[qi + 3 * HWSZ];
        }
#pragma unroll
        for (int di = 0; di < 9; ++di) {
#pragma unroll
            for (int dj = 0; dj < 9; ++dj) {
                const float4 k4 = *(const float4*)
                    &tile[((th + di) * HALO_W + (tw + dj)) * 4];
                float s = sims[di * 9 + dj];
                s = fmaf(q0, k4.x, s);
                s = fmaf(q1, k4.y, s);
                s = fmaf(q2, k4.z, s);
                s = fmaf(q3, k4.w, s);
                sims[di * 9 + dj] = s;
            }
        }
    }

    // ---------------- Softmax over 81 (registers) ----------------
    const float invs = 0.08838834764831845f;   // 1/sqrt(128)
    float mx = -3.0e38f;
#pragma unroll
    for (int o = 0; o < 81; ++o) {
        sims[o] *= invs;
        mx = fmaxf(mx, sims[o]);
    }
    float sum = 0.f;
#pragma unroll
    for (int o = 0; o < 81; ++o) {
        sims[o] = __expf(sims[o] - mx);
        sum += sims[o];
    }
    const float rs = 1.f / sum;
#pragma unroll
    for (int o = 0; o < 81; ++o) sims[o] *= rs;

    // ---------------- Pass 2: weighted sum of value ----------------
    for (int c0 = 0; c0 < CH; c0 += 4) {
        __syncthreads();
        for (int i = tid; i < TILE_E * 4; i += 256) {
            const int col = i % HALO_W;
            const int r   = (i / HALO_W) % HALO_H;
            const int cc  = i / (HALO_W * HALO_H);
            const int gh  = h0 + r - 4;
            const int gw  = w0 + col - 4;
            float vv = 0.f;
            if (gh >= 0 && gh < HH && gw >= 0 && gw < WW)
                vv = val[bbase + (size_t)(c0 + cc) * HWSZ + gh * WW + gw];
            tile[(r * HALO_W + col) * 4 + cc] = vv;
        }
        __syncthreads();

        float o0 = 0.f, o1 = 0.f, o2 = 0.f, o3 = 0.f;
#pragma unroll
        for (int di = 0; di < 9; ++di) {
#pragma unroll
            for (int dj = 0; dj < 9; ++dj) {
                const float4 v4 = *(const float4*)
                    &tile[((th + di) * HALO_W + (tw + dj)) * 4];
                const float wv = sims[di * 9 + dj];
                o0 = fmaf(wv, v4.x, o0);
                o1 = fmaf(wv, v4.y, o1);
                o2 = fmaf(wv, v4.z, o2);
                o3 = fmaf(wv, v4.w, o3);
            }
        }
        if (valid) {
            const size_t oi = bbase + (size_t)c0 * HWSZ + h * WW + w;
            out[oi]            = o0;
            out[oi + HWSZ]     = o1;
            out[oi + 2 * HWSZ] = o2;
            out[oi + 3 * HWSZ] = o3;
        }
    }
}

// ---------------------------------------------------------------------------
// Launch: 5 GEMM passes (q1->q2, k1->k2, v) then fused attention.
// Workspace layout (floats): [tmp][q][key][value], each 2*128*180*180.
// ---------------------------------------------------------------------------
extern "C" void kernel_launch(void* const* d_in, const int* in_sizes, int n_in,
                              void* d_out, int out_size, void* d_ws, size_t ws_size,
                              hipStream_t stream)
{
    const float* target = (const float*)d_in[0];
    const float* source = (const float*)d_in[1];

    // setup_inputs order: target, source, then {W,g,b,m,v} for q1,q2,k1,k2,v
    const float* P[5][5];
    for (int i = 0; i < 5; ++i)
        for (int j = 0; j < 5; ++j)
            P[i][j] = (const float*)d_in[2 + 5 * i + j];
    // i: 0=q1, 1=q2, 2=k1, 3=k2, 4=v

    const size_t TEN = (size_t)BATCH * CH * HWSZ;   // elements per tensor
    float* tmp  = (float*)d_ws;
    float* qb   = tmp + TEN;
    float* keyb = tmp + 2 * TEN;
    float* valb = tmp + 3 * TEN;

    const dim3 gB(BATCH * NTILES), gT(256);         // 4050 blocks, 8 waves ea.

    // q = relu(bn2(W2 @ relu(bn1(W1 @ target))))
    proj_gemm_kernel<<<gB, gT, 0, stream>>>(target, P[0][0], P[0][1], P[0][2], P[0][3], P[0][4], tmp);
    proj_gemm_kernel<<<gB, gT, 0, stream>>>(tmp,    P[1][0], P[1][1], P[1][2], P[1][3], P[1][4], qb);
    // key = relu(bn2(W2 @ relu(bn1(W1 @ source))))
    proj_gemm_kernel<<<gB, gT, 0, stream>>>(source, P[2][0], P[2][1], P[2][2], P[2][3], P[2][4], tmp);
    proj_gemm_kernel<<<gB, gT, 0, stream>>>(tmp,    P[3][0], P[3][1], P[3][2], P[3][3], P[3][4], keyb);
    // value = relu(bn(Wv @ source))
    proj_gemm_kernel<<<gB, gT, 0, stream>>>(source, P[4][0], P[4][1], P[4][2], P[4][3], P[4][4], valb);

    // fused local attention
    const dim3 aB((WW + TW - 1) / TW, HH / THT, BATCH);      // (3, 45, 2)
    local_attn_kernel<<<aB, dim3(256), 0, stream>>>(qb, keyb, valb, (float*)d_out);
}
